// MHA_38165079392270
// MI455X (gfx1250) — matrix operand order repaired and checked by reading, not compiled, not run on hardware
//
#include <hip/hip_runtime.h>

typedef _Float16 h16;
typedef __attribute__((ext_vector_type(16))) _Float16 v16h;
typedef __attribute__((ext_vector_type(8)))  float    v8f;

#define NEG_SLOPE 0.2f
#define EPS 1e-6f

// ---------- CDNA5 async global->LDS copy (ASYNCcnt path) ----------
__device__ __forceinline__ void async_ld_b128(unsigned lds_off, unsigned long long gaddr) {
  asm volatile("global_load_async_to_lds_b128 %0, %1, off"
               :: "v"(lds_off), "v"(gaddr)
               : "memory");
}
__device__ __forceinline__ void wait_async0() {
  asm volatile("s_wait_asynccnt 0" ::: "memory");
}

// ---------- WMMA fragment loaders (CDNA5 ISA 7.12.2 layouts, wave32) ----------

// A-matrix 16x32 f16 (MxK): lane m = lane&15, half = lane>>4.
// VGPR i (f16 pair 2i,2i+1): k = (i<4?0:16) + 2*(i&3) + half*8, elements k,k+1.
// Source: row-major [m][k] with leading dim ld (k-pairs contiguous).
__device__ __forceinline__ v16h frag_a_ld(const h16* base, int ld) {
  int lane = threadIdx.x & 31;
  int m = lane & 15, hf = lane >> 4;
  const h16* row = base + m * ld;
  v16h r;
#pragma unroll
  for (int i = 0; i < 8; ++i) {
    int k = ((i < 4) ? 0 : 16) + 2 * (i & 3) + hf * 8;
    r[2 * i]     = row[k];
    r[2 * i + 1] = row[k + 1];
  }
  return r;
}

// B-matrix 32x16 f16 (KxN): lane n = lane&15, half = lane>>4.
// VGPR j: k = (j<4?0:16) + 2*(j&3) + half*8, elements k,k+1.  Source indexed value(k,n)=base[n*ld+k]
// i.e. an n-major buffer with k contiguous (store the tile transposed).
__device__ __forceinline__ v16h frag_b_ld(const h16* base, int ld) {
  int lane = threadIdx.x & 31;
  int n = lane & 15, hf = lane >> 4;
  const h16* col = base + n * ld;
  v16h r;
#pragma unroll
  for (int j = 0; j < 8; ++j) {
    int k = ((j < 4) ? 0 : 16) + 2 * (j & 3) + hf * 8;
    r[2 * j]     = col[k];
    r[2 * j + 1] = col[k + 1];
  }
  return r;
}

// ---------- K0: zero ----------
__global__ void zero_kernel(float* __restrict__ p, int n) {
  int i = blockIdx.x * blockDim.x + threadIdx.x;
  if (i < n) p[i] = 0.f;
}

// ---------- K1: leaky-relu + QKV projection, h (f16, unnormalized) + sumsq ----------
// grid (16, 96): y = nbh = n*32 + b*8 + h ; x = T tile (128 rows)
__global__ __launch_bounds__(256) void qkv_proj_kernel(
    const float* __restrict__ x, const float* __restrict__ qkv,
    h16* __restrict__ hbuf, float* __restrict__ sumsq) {
  __shared__ h16 As[128 * 40];  // [t][dc] t-major, 32-d chunk, pad to 40
  __shared__ h16 Wt[64 * 40];   // [dh][dc] transposed weight chunk
  __shared__ float ss[64];

  int nbh = blockIdx.y;
  int n = nbh >> 5;
  int b = (nbh >> 3) & 3;
  int hh = nbh & 7;
  int t0 = blockIdx.x * 128;
  int tid = threadIdx.x;
  int wave = tid >> 5;
  int lane = tid & 31;
  int half = lane >> 4;
  int ncol = lane & 15;

  const float* xb = x + (size_t)b * 512 * 2048;
  const float* wb = qkv + (size_t)((n * 8 + hh) * 512) * 64;

  v8f acc[4] = {};

  for (int d0 = 0; d0 < 512; d0 += 32) {
    for (int idx = tid; idx < 32 * 128; idx += 256) {
      int dc = idx >> 7, tt = idx & 127;
      float v = xb[(size_t)(d0 + dc) * 2048 + t0 + tt];
      v = v >= 0.f ? v : NEG_SLOPE * v;               // leaky relu
      As[tt * 40 + dc] = (h16)v;
    }
    for (int idx = tid; idx < 32 * 64; idx += 256) {
      int dc = idx >> 6, dh = idx & 63;
      Wt[dh * 40 + dc] = (h16)wb[(size_t)(d0 + dc) * 64 + dh];
    }
    __syncthreads();
    v16h a = frag_a_ld(As + wave * 16 * 40, 40);
#pragma unroll
    for (int t = 0; t < 4; ++t) {
      v16h bf = frag_b_ld(Wt + t * 16 * 40, 40);
      acc[t] = __builtin_amdgcn_wmma_f32_16x16x32_f16(
          false, a, false, bf, (short)0, acc[t], false, false);
    }
    __syncthreads();
  }

  if (tid < 64) ss[tid] = 0.f;
  __syncthreads();

  h16* hrow0 = hbuf + (size_t)nbh * 2048 * 64 + (size_t)(t0 + wave * 16) * 64;
#pragma unroll
  for (int t = 0; t < 4; ++t) {
    float s2 = 0.f;
#pragma unroll
    for (int r = 0; r < 8; ++r) {
      int m = r + half * 8;                            // C/D layout: VGPR r -> row r/r+8
      float v = acc[t][r];
      hrow0[(size_t)m * 64 + t * 16 + ncol] = (h16)v;
      s2 += v * v;
    }
    atomicAdd(&ss[t * 16 + ncol], s2);                 // ds_add_f32
  }
  __syncthreads();
  if (tid < 64) atomicAdd(&sumsq[nbh * 64 + tid], ss[tid]);
}

// ---------- K2: per-channel RMSNorm scale ----------
__global__ void scale_kernel(const float* __restrict__ sumsq,
                             const float* __restrict__ norm_w,
                             float* __restrict__ scale, int n) {
  int i = blockIdx.x * blockDim.x + threadIdx.x;
  if (i < n) scale[i] = norm_w[0] * rsqrtf(sumsq[i] * (1.f / 2048.f) + EPS);
}

// ---------- K2b: apply scale in place to f16 q/k/v ----------
__global__ void norm_kernel(h16* __restrict__ hbuf, const float* __restrict__ scale) {
  size_t i = (size_t)blockIdx.x * 256 + threadIdx.x;   // 12,582,912 total
  int chan = (int)(i >> 17) * 64 + (int)(i & 63);      // 2048*64 = 2^17
  hbuf[i] = (h16)((float)hbuf[i] * scale[chan]);
}

// ---------- K3: flash attention with transposed output ----------
// grid (16, 32): y = bh = b*8 + h ; x = 128-query block. 256 thr = 8 waves,
// each wave owns 16 query rows.
__global__ __launch_bounds__(256) void attn_kernel(
    const h16* __restrict__ hbuf, float* __restrict__ out) {
  __shared__ h16 Qs[128 * 72];
  __shared__ h16 Kt[64 * 72];       // [key][dh]
  __shared__ h16 Vt[64 * 72];       // [dh][key] (transposed)
  __shared__ h16 Ps[8 * 16 * 72];   // per-wave P tile [m][key]

  int bh = blockIdx.y;
  int b = bh >> 3, hh = bh & 7;
  int qt0 = blockIdx.x * 128;
  int tid = threadIdx.x, wave = tid >> 5, lane = tid & 31;
  int half = lane >> 4, ncol = lane & 15;

  const size_t HS = (size_t)2048 * 64;
  const h16* qf = hbuf + (size_t)(0 * 32 + bh) * HS;
  const h16* kf = hbuf + (size_t)(1 * 32 + bh) * HS;
  const h16* vf = hbuf + (size_t)(2 * 32 + bh) * HS;

  // stage Q tile: 128 rows x 64 f16 = 1024 x 16B chunks, async direct-to-LDS
  {
    unsigned qbase = (unsigned)(size_t)(void*)Qs;
    for (int c = tid; c < 128 * 8; c += 256) {
      int row = c >> 3, seg = c & 7;
      async_ld_b128(qbase + (unsigned)(row * 72 + seg * 8) * 2,
                    (unsigned long long)(size_t)(qf + (size_t)(qt0 + row) * 64 + seg * 8));
    }
    wait_async0();
  }
  __syncthreads();
  v16h qa[2];
  qa[0] = frag_a_ld(Qs + wave * 16 * 72, 72);
  qa[1] = frag_a_ld(Qs + wave * 16 * 72 + 32, 72);

  v8f oacc[4] = {};
  float mrow[8], lrow[8];
#pragma unroll
  for (int r = 0; r < 8; ++r) { mrow[r] = -1e30f; lrow[r] = 0.f; }

  h16* Pw = Ps + wave * 16 * 72;
  unsigned kbase = (unsigned)(size_t)(void*)Kt;

  for (int j = 0; j < 2048; j += 64) {
    __syncthreads();
    // K tile: 64 rows x 64 f16 = 512 x 16B chunks, async direct-to-LDS
    for (int c = tid; c < 64 * 8; c += 256) {
      int row = c >> 3, seg = c & 7;
      async_ld_b128(kbase + (unsigned)(row * 72 + seg * 8) * 2,
                    (unsigned long long)(size_t)(kf + (size_t)(j + row) * 64 + seg * 8));
    }
    // V tile transposed (element-wise, regular ds path)
    for (int idx = tid; idx < 64 * 64; idx += 256) {
      int key = idx >> 6, dh = idx & 63;
      Vt[dh * 72 + key] = vf[(size_t)(j + key) * 64 + dh];
    }
    if (j + 64 < 2048) {                                // prefetch next tiles
      __builtin_prefetch(kf + (size_t)(j + 64) * 64 + tid, 0, 1);
      __builtin_prefetch(vf + (size_t)(j + 64) * 64 + tid, 0, 1);
    }
    wait_async0();
    __syncthreads();

    // S = Q K^T  (16 x 64 per wave)
    v8f s[4] = {};
#pragma unroll
    for (int kc = 0; kc < 2; ++kc) {
#pragma unroll
      for (int t = 0; t < 4; ++t) {
        v16h bf = frag_b_ld(Kt + t * 16 * 72 + kc * 32, 72);
        s[t] = __builtin_amdgcn_wmma_f32_16x16x32_f16(
            false, qa[kc], false, bf, (short)0, s[t], false, false);
      }
    }
#pragma unroll
    for (int t = 0; t < 4; ++t)
#pragma unroll
      for (int r = 0; r < 8; ++r) s[t][r] *= 0.125f;    // 1/sqrt(64)

    // online softmax: row max over 64 keys
    float rmax[8];
#pragma unroll
    for (int r = 0; r < 8; ++r)
      rmax[r] = fmaxf(fmaxf(s[0][r], s[1][r]), fmaxf(s[2][r], s[3][r]));
#pragma unroll
    for (int msk = 1; msk <= 8; msk <<= 1)
#pragma unroll
      for (int r = 0; r < 8; ++r)
        rmax[r] = fmaxf(rmax[r], __shfl_xor(rmax[r], msk, 32));

    float alpha[8], mnew[8], rsum[8];
#pragma unroll
    for (int r = 0; r < 8; ++r) {
      mnew[r] = fmaxf(mrow[r], rmax[r]);
      alpha[r] = __expf(mrow[r] - mnew[r]);
      mrow[r] = mnew[r];
      rsum[r] = 0.f;
    }

#pragma unroll
    for (int t = 0; t < 4; ++t)
#pragma unroll
      for (int r = 0; r < 8; ++r) {
        float p = __expf(s[t][r] - mnew[r]);
        rsum[r] += p;
        int m = r + half * 8;
        Pw[m * 72 + t * 16 + ncol] = (h16)p;            // C-layout -> LDS -> A-layout
      }
#pragma unroll
    for (int msk = 1; msk <= 8; msk <<= 1)
#pragma unroll
      for (int r = 0; r < 8; ++r) rsum[r] += __shfl_xor(rsum[r], msk, 32);
#pragma unroll
    for (int r = 0; r < 8; ++r) lrow[r] = lrow[r] * alpha[r] + rsum[r];

#pragma unroll
    for (int t = 0; t < 4; ++t)
#pragma unroll
      for (int r = 0; r < 8; ++r) oacc[t][r] *= alpha[r];

    // O += P V
#pragma unroll
    for (int kc = 0; kc < 2; ++kc) {
      v16h pa = frag_a_ld(Pw + kc * 32, 72);
#pragma unroll
      for (int t = 0; t < 4; ++t) {
        v16h bf = frag_b_ld(Vt + t * 16 * 72 + kc * 32, 72);
        oacc[t] = __builtin_amdgcn_wmma_f32_16x16x32_f16(
            false, pa, false, bf, (short)0, oacc[t], false, false);
      }
    }
  }

  // out[(b*512 + hh*64 + dh) * 2048 + t]  (transposed store)
  float* ob = out + ((size_t)b * 512 + (size_t)hh * 64) * 2048;
#pragma unroll
  for (int t = 0; t < 4; ++t)
#pragma unroll
    for (int r = 0; r < 8; ++r) {
      int m = r + half * 8;
      int dh = t * 16 + ncol;
      int tq = qt0 + wave * 16 + m;
      ob[(size_t)dh * 2048 + tq] = oacc[t][r] / lrow[r];
    }
}

// ---------- launch ----------
extern "C" void kernel_launch(void* const* d_in, const int* in_sizes, int n_in,
                              void* d_out, int out_size, void* d_ws, size_t ws_size,
                              hipStream_t stream) {
  const float* x      = (const float*)d_in[0];  // (4,512,2048)
  const float* qkv    = (const float*)d_in[1];  // (3,8,512,64)
  const float* norm_w = (const float*)d_in[2];  // (1,)
  float* out = (float*)d_out;                   // (4,512,2048)

  // workspace layout: f16 h (3*4*8*2048*64 = 12,582,912 halves = 25,165,824 B)
  // then sumsq[6144] f32, then scale[6144] f32.
  h16*   hbuf     = (h16*)d_ws;
  float* sumsq    = (float*)((char*)d_ws + 25165824u);
  float* scalebuf = sumsq + 6144;

  zero_kernel<<<24, 256, 0, stream>>>(sumsq, 6144);
  qkv_proj_kernel<<<dim3(16, 96), 256, 0, stream>>>(x, qkv, hbuf, sumsq);
  scale_kernel<<<24, 256, 0, stream>>>(sumsq, norm_w, scalebuf, 6144);
  norm_kernel<<<49152, 256, 0, stream>>>(hbuf, scalebuf);
  attn_kernel<<<dim3(16, 32), 256, 0, stream>>>(hbuf, out);
}
